// Spatial_Attention_70385924047394
// MI455X (gfx1250) — compile-verified, matrix operand order checked by reading, not run on hardware
//
#include <hip/hip_runtime.h>
#include <hip/hip_bf16.h>
#include <math.h>

#define B_ 4
#define T_ 24
#define N_ 1024
#define F_ 128
#define M_ (B_*T_*N_)          // 98304 rows
#define FRONT_T_ 12
#define SCALE 0.08838834764831845f   // 1/sqrt(128)
#define EPS_ 1e-5f

typedef __attribute__((ext_vector_type(16))) _Float16 v16h;
typedef __attribute__((ext_vector_type(8)))  _Float16 h8v;
typedef __attribute__((ext_vector_type(8)))  float    v8f;

// Load a 16x32 f16 WMMA fragment from an LDS row-major matrix.
// A operand: row0+lane%16 selects the M row; halves 0-7 cover k0+8*(lane/16)..+8,
// halves 8-15 cover k0+16+8*(lane/16)..+8 (per CDNA5 16-bit A layout).
// B operand is loaded from B^T stored row-major (column of B contiguous), which
// mirrors the A layout under transpose.
__device__ __forceinline__ v16h frag_ld(const _Float16* base, int stride, int row0, int k0) {
  const int lane = threadIdx.x & 31;
  const _Float16* p = base + (row0 + (lane & 15)) * stride + k0 + ((lane >> 4) << 3);
  union { v16h v; h8v h[2]; } u;
  u.h[0] = *(const h8v*)p;
  u.h[1] = *(const h8v*)(p + 16);
  return u.v;
}

__device__ __forceinline__ v8f wmma16(v16h a, v16h b, v8f c) {
  return __builtin_amdgcn_wmma_f32_16x16x32_f16(false, a, false, b, (short)0, c, false, false);
}

// ---------------- Kernel 1: QKV projection ----------------
// out[row][o] = sum_k x[row][k]*W_xff[o][k] + b_xff[o], split into Q/K/V chunks.
__global__ void __launch_bounds__(256) k_qkv(const float* __restrict__ x,
    const float* __restrict__ Wx, const float* __restrict__ bx,
    float* __restrict__ Q, float* __restrict__ K, float* __restrict__ V) {
  __shared__ __align__(16) _Float16 Xs[64*128];
  __shared__ __align__(16) _Float16 Ws[128*128];
  const int rb = blockIdx.x * 64;
  for (int i = threadIdx.x; i < 64*128; i += 256) Xs[i] = (_Float16)x[(size_t)rb*F_ + i];
  const int w = threadIdx.x >> 5, lane = threadIdx.x & 31;
  const int wr = w & 3, wc = w >> 2, half = lane >> 4, col0 = lane & 15;
  float* outs[3] = {Q, K, V};
  for (int cc = 0; cc < 3; ++cc) {
    __syncthreads();
    for (int i = threadIdx.x; i < 128*128; i += 256) Ws[i] = (_Float16)Wx[cc*128*128 + i];
    __syncthreads();
    v8f acc[4] = {};
    for (int ks = 0; ks < 4; ++ks) {
      v16h a = frag_ld(Xs, 128, wr*16, ks*32);
      for (int tn = 0; tn < 4; ++tn) {
        v16h bb = frag_ld(Ws, 128, wc*64 + tn*16, ks*32);
        acc[tn] = wmma16(a, bb, acc[tn]);
      }
    }
    float* O = outs[cc];
    for (int tn = 0; tn < 4; ++tn) {
      const int o = wc*64 + tn*16 + col0;
      const float bias = bx[cc*128 + o];
      for (int r = 0; r < 8; ++r) {
        const int m = wr*16 + r + half*8;
        O[(size_t)(rb + m)*F_ + o] = acc[tn][r] + bias;
      }
    }
  }
}

// ---------------- Kernel 2: flash attention per (b,t) ----------------
__global__ void __launch_bounds__(256) k_attn(
    const float* __restrict__ Q, const float* __restrict__ K, const float* __restrict__ V,
    float* __restrict__ O, float* __restrict__ mstat, float* __restrict__ lstat) {
  __shared__ __align__(16) _Float16 Qs[64*128];
  __shared__ __align__(16) _Float16 Ks[64*128];  // overlaid by P (8KB) + alpha + l after S GEMM
  __shared__ __align__(16) _Float16 Vt[128*64];  // V tile transposed: [f][keyrow]
  __shared__ __align__(16) float    Ss[64*64];
  _Float16* Ps      = Ks;
  float*    alphaL  = (float*)(Ks + 64*64);
  float*    lL      = alphaL + 64;

  const int bt = blockIdx.x >> 4;
  const int qb = blockIdx.x & 15;
  const float* Qg = Q + (size_t)bt*N_*F_;
  const float* Kg = K + (size_t)bt*N_*F_;
  const float* Vg = V + (size_t)bt*N_*F_;
  const int w = threadIdx.x >> 5, lane = threadIdx.x & 31;
  const int wr = w & 3, wc = w >> 2, half = lane >> 4, col0 = lane & 15;

  for (int i = threadIdx.x; i < 64*128; i += 256) Qs[i] = (_Float16)Qg[(size_t)(qb*64)*F_ + i];

  float m_r = -INFINITY, l_r = 0.0f;
  v8f acc[4] = {};

  for (int kb = 0; kb < 16; ++kb) {
    __syncthreads();
    for (int i = threadIdx.x; i < 64*128; i += 256) Ks[i] = (_Float16)Kg[(size_t)(kb*64)*F_ + i];
    for (int i = threadIdx.x; i < 64*128; i += 256) {
      const int vr = i >> 7, f = i & 127;
      Vt[f*64 + vr] = (_Float16)Vg[(size_t)(kb*64 + vr)*F_ + f];
    }
    __syncthreads();
    if (w < 4) {                 // waves 0-3: S = Q Kt, 16 rows x 64 cols each
      v8f sacc[4] = {};
      for (int ks = 0; ks < 4; ++ks) {
        v16h a = frag_ld(Qs, 128, w*16, ks*32);
        for (int tn = 0; tn < 4; ++tn) {
          v16h bb = frag_ld(Ks, 128, tn*16, ks*32);
          sacc[tn] = wmma16(a, bb, sacc[tn]);
        }
      }
      for (int tn = 0; tn < 4; ++tn)
        for (int r = 0; r < 8; ++r)
          Ss[(w*16 + r + half*8)*64 + tn*16 + col0] = sacc[tn][r] * SCALE;
    }
    __syncthreads();
    if (threadIdx.x < 64) {      // online softmax, one thread per query row
      const int r = threadIdx.x;
      float mn = m_r;
      for (int c = 0; c < 64; ++c) mn = fmaxf(mn, Ss[r*64 + c]);
      const float al = __expf(m_r - mn);
      float ln = l_r * al;
      for (int c = 0; c < 64; ++c) {
        const float p = __expf(Ss[r*64 + c] - mn);
        ln += p;
        Ps[r*64 + c] = (_Float16)p;
      }
      m_r = mn; l_r = ln;
      alphaL[r] = al;
    }
    __syncthreads();
    for (int r = 0; r < 8; ++r) {                  // rescale accumulators
      const float a = alphaL[wr*16 + r + half*8];
      acc[0][r] *= a; acc[1][r] *= a; acc[2][r] *= a; acc[3][r] *= a;
    }
    for (int ks = 0; ks < 2; ++ks) {               // O += P @ V
      v16h a = frag_ld(Ps, 64, wr*16, ks*32);
      for (int tn = 0; tn < 4; ++tn) {
        v16h bb = frag_ld(Vt, 64, wc*64 + tn*16, ks*32);
        acc[tn] = wmma16(a, bb, acc[tn]);
      }
    }
  }
  __syncthreads();
  if (threadIdx.x < 64) {
    lL[threadIdx.x] = l_r;
    const int n = qb*64 + threadIdx.x;
    mstat[(size_t)bt*N_ + n] = m_r;
    lstat[(size_t)bt*N_ + n] = l_r;
  }
  __syncthreads();
  for (int tn = 0; tn < 4; ++tn)
    for (int r = 0; r < 8; ++r) {
      const int m = wr*16 + r + half*8;
      const int f = wc*64 + tn*16 + col0;
      O[((size_t)bt*N_ + qb*64 + m)*F_ + f] = acc[tn][r] / lL[m];
    }
}

// ---------------- Kernel 3a: softmax column sums at t = T-1 ----------------
__global__ void __launch_bounds__(256) k_colsum(
    const float* __restrict__ Q, const float* __restrict__ K,
    const float* __restrict__ mstat, const float* __restrict__ lstat,
    float* __restrict__ colsum) {
  __shared__ __align__(16) _Float16 Kc[64*128];
  __shared__ __align__(16) _Float16 Qb[64*128];
  __shared__ __align__(16) float    Ss[64*64];
  const int b  = blockIdx.x >> 4;
  const int cb = blockIdx.x & 15;
  const int bt = b*T_ + (T_-1);
  const float* Qg = Q + (size_t)bt*N_*F_;
  const float* Kg = K + (size_t)bt*N_*F_;
  const float* ms = mstat + (size_t)bt*N_;
  const float* ls = lstat + (size_t)bt*N_;
  const int w = threadIdx.x >> 5, lane = threadIdx.x & 31;
  const int half = lane >> 4, col0 = lane & 15;
  const int mcol = threadIdx.x & 63, quad = threadIdx.x >> 6;

  for (int i = threadIdx.x; i < 64*128; i += 256) Kc[i] = (_Float16)Kg[(size_t)(cb*64)*F_ + i];
  float part = 0.0f;
  for (int nb = 0; nb < 16; ++nb) {
    __syncthreads();
    for (int i = threadIdx.x; i < 64*128; i += 256) Qb[i] = (_Float16)Qg[(size_t)(nb*64)*F_ + i];
    __syncthreads();
    if (w < 4) {
      v8f sacc[4] = {};
      for (int ks = 0; ks < 4; ++ks) {
        v16h a = frag_ld(Qb, 128, w*16, ks*32);
        for (int tn = 0; tn < 4; ++tn) {
          v16h bb = frag_ld(Kc, 128, tn*16, ks*32);
          sacc[tn] = wmma16(a, bb, sacc[tn]);
        }
      }
      for (int tn = 0; tn < 4; ++tn)
        for (int r = 0; r < 8; ++r)
          Ss[(w*16 + r + half*8)*64 + tn*16 + col0] = sacc[tn][r] * SCALE;
    }
    __syncthreads();
    for (int rr = 0; rr < 16; ++rr) {
      const int rl = quad*16 + rr;
      const int n  = nb*64 + rl;
      part += __expf(Ss[rl*64 + mcol] - ms[n]) / ls[n];
    }
  }
  __syncthreads();
  Ss[quad*64 + mcol] = part;
  __syncthreads();
  if (threadIdx.x < 64)
    colsum[b*N_ + cb*64 + threadIdx.x] =
      Ss[threadIdx.x] + Ss[64 + threadIdx.x] + Ss[128 + threadIdx.x] + Ss[192 + threadIdx.x];
}

// ---------------- Kernel 3b: 3rd-smallest column per batch ----------------
__global__ void k_select(const float* __restrict__ colsum, int* __restrict__ ysel) {
  const int b = threadIdx.x;
  if (b >= B_) return;
  const float* cs = colsum + b*N_;
  int e0 = -1, e1 = -1, best = 0;
  for (int pass = 0; pass < 3; ++pass) {
    float bv = INFINITY; int bi = 0;
    for (int i = 0; i < N_; ++i) {
      if (i == e0 || i == e1) continue;
      const float v = cs[i];
      if (v < bv) { bv = v; bi = i; }   // ties -> lowest index, matching top_k
    }
    if (pass == 0) e0 = bi; else if (pass == 1) e1 = bi;
    best = bi;
  }
  ysel[b] = best;
}

// ---------------- Kernel 4: uint8-quantized mix row + rank-1 fixup ----------------
__global__ void __launch_bounds__(256) k_mixfix(
    const float* __restrict__ Q, const float* __restrict__ K, const float* __restrict__ V,
    const float* __restrict__ mstat, const float* __restrict__ lstat,
    const int* __restrict__ ysel, float* __restrict__ O) {
  __shared__ float delta[128];
  __shared__ float ky[128];
  __shared__ float pL[1024];
  const int b = blockIdx.x / FRONT_T_;
  const int t = blockIdx.x % FRONT_T_;
  const int bt = b*T_ + t;
  const size_t base = (size_t)bt*N_*F_;
  const int y  = ysel[b];
  const int ys = (y-1 > 0) ? (y-1) : 0;
  const int ye = (y+2 < N_) ? (y+2) : N_;
  const float cnt = (float)(ye - ys);
  if (threadIdx.x < 128) {
    const int f = threadIdx.x;
    float s = 0.0f;
    for (int r = ys; r < ye; ++r) s += V[base + (size_t)r*F_ + f];
    float mixed = s / cnt;
    float q8;                                 // float -> uint8 (saturating trunc)
    if (mixed <= 0.0f) q8 = 0.0f;
    else if (mixed >= 255.0f) q8 = 255.0f;
    else q8 = (float)(unsigned int)mixed;
    delta[f] = q8 - V[base + (size_t)y*F_ + f];
    ky[f]    = K[base + (size_t)y*F_ + f];
  }
  __syncthreads();
  for (int n = threadIdx.x; n < N_; n += 256) {
    float d = 0.0f;
    const float* qr = Q + base + (size_t)n*F_;
    for (int f = 0; f < F_; ++f) d += qr[f] * ky[f];
    pL[n] = __expf(d*SCALE - mstat[(size_t)bt*N_ + n]) / lstat[(size_t)bt*N_ + n];
  }
  __syncthreads();
  const int f  = threadIdx.x & 127;
  const int h0 = threadIdx.x >> 7;
  const float d = delta[f];
  for (int n = h0; n < N_; n += 2)
    O[base + (size_t)n*F_ + f] += pL[n] * d;
}

// ---------------- Kernel 5: FFN + residual + LayerNorm ----------------
__global__ void __launch_bounds__(256) k_ffn_ln(
    const float* __restrict__ x,
    const float* __restrict__ W1, const float* __restrict__ b1,
    const float* __restrict__ W2, const float* __restrict__ b2,
    const float* __restrict__ g,  const float* __restrict__ be,
    float* __restrict__ O) {
  __shared__ __align__(16) _Float16 At[64*128];     // value_i tile, then H (gelu) tile
  __shared__ __align__(16) _Float16 Ws[128*128];    // W1, then W2, then res (f32 overlay)
  float* res = (float*)Ws;
  const int rb = blockIdx.x * 64;
  const int w = threadIdx.x >> 5, lane = threadIdx.x & 31;
  const int wr = w & 3, wc = w >> 2, half = lane >> 4, col0 = lane & 15;

  for (int i = threadIdx.x; i < 64*128;  i += 256) At[i] = (_Float16)O[(size_t)rb*F_ + i];
  for (int i = threadIdx.x; i < 128*128; i += 256) Ws[i] = (_Float16)W1[i];
  __syncthreads();
  v8f acc[4] = {};
  for (int ks = 0; ks < 4; ++ks) {
    v16h a = frag_ld(At, 128, wr*16, ks*32);
    for (int tn = 0; tn < 4; ++tn) {
      v16h bb = frag_ld(Ws, 128, wc*64 + tn*16, ks*32);
      acc[tn] = wmma16(a, bb, acc[tn]);
    }
  }
  __syncthreads();
  for (int tn = 0; tn < 4; ++tn) {                  // gelu -> H tile (reuse At)
    const int o = wc*64 + tn*16 + col0;
    const float bias = b1[o];
    for (int r = 0; r < 8; ++r) {
      const int m = wr*16 + r + half*8;
      const float v = acc[tn][r] + bias;
      At[m*128 + o] = (_Float16)(0.5f * v * (1.0f + erff(v * 0.70710678118654752f)));
    }
  }
  for (int i = threadIdx.x; i < 128*128; i += 256) Ws[i] = (_Float16)W2[i];
  __syncthreads();
  v8f acc2[4] = {};
  for (int ks = 0; ks < 4; ++ks) {
    v16h a = frag_ld(At, 128, wr*16, ks*32);
    for (int tn = 0; tn < 4; ++tn) {
      v16h bb = frag_ld(Ws, 128, wc*64 + tn*16, ks*32);
      acc2[tn] = wmma16(a, bb, acc2[tn]);
    }
  }
  __syncthreads();
  for (int tn = 0; tn < 4; ++tn) {                  // residual into f32 LDS
    const int o = wc*64 + tn*16 + col0;
    const float bias = b2[o];
    for (int r = 0; r < 8; ++r) {
      const int m = wr*16 + r + half*8;
      res[m*128 + o] = acc2[tn][r] + bias + x[(size_t)(rb + m)*F_ + o];
    }
  }
  __syncthreads();
  if (threadIdx.x < 64) {                           // LayerNorm over F=128
    const int m = threadIdx.x;
    float mu = 0.0f;
    for (int o = 0; o < 128; ++o) mu += res[m*128 + o];
    mu *= (1.0f/128.0f);
    float var = 0.0f;
    for (int o = 0; o < 128; ++o) { const float dd = res[m*128 + o] - mu; var += dd*dd; }
    var *= (1.0f/128.0f);
    const float rs = rsqrtf(var + EPS_);
    for (int o = 0; o < 128; ++o)
      O[(size_t)(rb + m)*F_ + o] = (res[m*128 + o] - mu)*rs*g[o] + be[o];
  }
}

extern "C" void kernel_launch(void* const* d_in, const int* in_sizes, int n_in,
                              void* d_out, int out_size, void* d_ws, size_t ws_size,
                              hipStream_t stream) {
  const float* x  = (const float*)d_in[0];
  const float* Wx = (const float*)d_in[1];
  const float* bx = (const float*)d_in[2];
  const float* W1 = (const float*)d_in[3];
  const float* b1 = (const float*)d_in[4];
  const float* W2 = (const float*)d_in[5];
  const float* b2 = (const float*)d_in[6];
  const float* g  = (const float*)d_in[7];
  const float* be = (const float*)d_in[8];
  float* out = (float*)d_out;

  float* ws = (float*)d_ws;
  const size_t mf = (size_t)M_ * F_;
  float* Qd = ws;                 // 48 MB
  float* Kd = Qd + mf;            // 48 MB
  float* Vd = Kd + mf;            // 48 MB
  float* ms = Vd + mf;            // rowmax  (B*T*N)
  float* ls = ms + M_;            // rowsum  (B*T*N)
  float* cs = ls + M_;            // colsums (B*N)
  int*   ys = (int*)(cs + B_*N_); // selected column per batch

  k_qkv   <<<M_/64,        256, 0, stream>>>(x, Wx, bx, Qd, Kd, Vd);
  k_attn  <<<M_/64,        256, 0, stream>>>(Qd, Kd, Vd, out, ms, ls);
  k_colsum<<<B_*16,        256, 0, stream>>>(Qd, Kd, ms, ls, cs);
  k_select<<<1, 32,           0, stream>>>(cs, ys);
  k_mixfix<<<B_*FRONT_T_,  256, 0, stream>>>(Qd, Kd, Vd, ms, ls, ys, out);
  k_ffn_ln<<<M_/64,        256, 0, stream>>>(x, W1, b1, W2, b2, g, be, out);
}
